// SimpleToxicityGNN_5179730559201
// MI455X (gfx1250) — matrix operand order repaired and checked by reading, not compiled, not run on hardware
//
#include <hip/hip_runtime.h>
#include <hip/hip_bf16.h>
#include <stdint.h>

#define D 128

typedef __attribute__((ext_vector_type(16))) __bf16 v16bf;
typedef __attribute__((ext_vector_type(8)))  __bf16 v8bf;
typedef __attribute__((ext_vector_type(8)))  float  v8f;

static __device__ __forceinline__ __bf16 f2bf(float f) {
  // round-to-nearest-even f32 -> bf16
  unsigned u = __float_as_uint(f);
  u += 0x7FFFu + ((u >> 16) & 1u);
  unsigned short s = (unsigned short)(u >> 16);
  return __builtin_bit_cast(__bf16, s);
}

// ---------------- degree ----------------
__global__ void init_deg_kernel(float* deg, long long N) {
  long long i = (long long)blockIdx.x * blockDim.x + threadIdx.x;
  if (i < N) deg[i] = 1.0f;
}

__global__ void deg_accum_kernel(const long long* __restrict__ dst, float* deg, long long E) {
  long long e = (long long)blockIdx.x * blockDim.x + threadIdx.x;
  if (e < E) unsafeAtomicAdd(&deg[dst[e]], 1.0f);
}

__global__ void deg_fin_kernel(const float* __restrict__ deg, float* dis, float* dinv, long long N) {
  long long i = (long long)blockIdx.x * blockDim.x + threadIdx.x;
  if (i < N) {
    float d = deg[i];
    dis[i]  = rsqrtf(d);
    dinv[i] = 1.0f / d;
  }
}

// ---------------- generic f32 -> bf16 row-major copy ----------------
__global__ void f32_to_bf16_kernel(const float* __restrict__ src, __bf16* __restrict__ dstb,
                                   long long n) {
  long long i = (long long)blockIdx.x * blockDim.x + threadIdx.x;
  if (i < n) dstb[i] = f2bf(src[i]);
}

// ---------------- weight pack: f32 [128,128] -> WMMA-B lane layout ----------------
// Wp[(((ct*4)+ks)*32 + lane)*16 + j] = bf16( W[(ks*32 + (j>>3)*16 + (lane>>4)*8 + (j&7))*128 + ct*16 + (lane&15)] )
// so each lane's 16-element B fragment (two K-runs of 8, matching A) is 32 contiguous bytes.
__global__ void pack_w_kernel(const float* __restrict__ w, __bf16* __restrict__ wp) {
  int idx = blockIdx.x * blockDim.x + threadIdx.x;   // 0 .. 16383
  if (idx >= D * D) return;
  int j    = idx & 15;
  int lane = (idx >> 4) & 31;
  int ks   = (idx >> 9) & 3;
  int ct   = idx >> 11;
  int krow = ks * 32 + ((j >> 3) << 4) + ((lane >> 4) << 3) + (j & 7);
  int col  = ct * 16 + (lane & 15);
  wp[idx] = f2bf(w[krow * D + col]);
}

// ---------------- WMMA GEMM: Out[M,128] = A_bf16[M,128] @ Wp ----------------
// One block = 8 waves = 16-row slab; wave w owns the 16x16 tile at cols [16w,16w+16).
__global__ __launch_bounds__(256)
void wmma_gemm_kernel(const __bf16* __restrict__ A, const __bf16* __restrict__ Bp,
                      float* __restrict__ Out, long long M) {
  const int wave = threadIdx.x >> 5;
  const int lane = threadIdx.x & 31;
  const long long row0 = (long long)blockIdx.x * 16;
  const int col0 = wave * 16;

  // A layout (16-bit A 16x32): lane holds row M=lane&15; its 16 elems are two
  // contiguous K-runs of 8: [kg..kg+7] and [16+kg..16+kg+7], kg = 8*(lane>=16)
  const int rA  = lane & 15;
  const int kgA = (lane >> 4) << 3;   // 0 or 8

  long long rowA = row0 + rA;
  if (rowA > M - 1) rowA = M - 1;     // clamp tail reads; keep EXEC all-ones for WMMA
  const __bf16* arow = A + rowA * D;
  const __bf16* bp   = Bp + ((size_t)(wave * 4) * 32 + lane) * 16;

  v8f acc = {};
#pragma unroll
  for (int ks = 0; ks < 4; ++ks) {
    const int k0 = ks * 32;
    v8bf a_lo = *(const v8bf*)(arow + k0 + kgA);        // 16B aligned
    v8bf a_hi = *(const v8bf*)(arow + k0 + 16 + kgA);   // 16B aligned
    v16bf a = __builtin_shufflevector(a_lo, a_hi,
                                      0, 1, 2, 3, 4, 5, 6, 7,
                                      8, 9, 10, 11, 12, 13, 14, 15);
    v16bf b = *(const v16bf*)(bp + (size_t)ks * 32 * 16);  // 32B contiguous
    acc = __builtin_amdgcn_wmma_f32_16x16x32_bf16(false, a, false, b, (short)0, acc,
                                                  false, false);
  }

  // C/D layout: VGPR r -> (M = r + 8*(lane>=16), N = lane&15)
  const int n = lane & 15;
  const int mbase = (lane >> 4) * 8;
  if (row0 + 16 <= M) {
    // full block: branch-free clause of stores at fixed offsets
    float* orow = Out + (row0 + mbase) * D + col0 + n;
#pragma unroll
    for (int r = 0; r < 8; ++r) orow[(size_t)r * D] = acc[r];
  } else {
    // tail block (head GEMM only): guarded stores
#pragma unroll
    for (int r = 0; r < 8; ++r) {
      long long m = row0 + mbase + r;
      if (m < M) Out[m * D + col0 + n] = acc[r];
    }
  }
}

// ---------------- agg init with self-loop term: agg = xw*deg_inv[row] + bias[col] ----------------
__global__ void init_agg_kernel(const float* __restrict__ xw, const float* __restrict__ dinv,
                                const float* __restrict__ bias, float* __restrict__ agg,
                                long long total) {
  long long i = (long long)blockIdx.x * blockDim.x + threadIdx.x;
  if (i < total) {
    long long row = i >> 7;
    int col = (int)(i & (D - 1));
    agg[i] = xw[i] * dinv[row] + bias[col];
  }
}

// ---------------- edge scatter: agg[dst] += xw[src] * (dis[src]*dis[dst]) ----------------
// one wave per edge; each lane handles 4 features (float4)
__global__ __launch_bounds__(256)
void scatter_kernel(const float* __restrict__ xw, const long long* __restrict__ src,
                    const long long* __restrict__ dst, const float* __restrict__ dis,
                    float* __restrict__ agg, long long E) {
  long long t = (long long)blockIdx.x * blockDim.x + threadIdx.x;
  long long e = t >> 5;
  if (e >= E) return;
  int lane = (int)(t & 31);
  long long s = src[e], d = dst[e];
  float norm = dis[s] * dis[d];
  const float4 v = ((const float4*)(xw + s * D))[lane];
  float* ar = agg + d * D + lane * 4;
  unsafeAtomicAdd(ar + 0, v.x * norm);
  unsafeAtomicAdd(ar + 1, v.y * norm);
  unsafeAtomicAdd(ar + 2, v.z * norm);
  unsafeAtomicAdd(ar + 3, v.w * norm);
}

// ReLU in place on f32 h, plus bf16 shadow copy for the next layer's GEMM
__global__ void relu_bf_kernel(float* h, __bf16* __restrict__ hb, long long total) {
  long long i = (long long)blockIdx.x * blockDim.x + threadIdx.x;
  if (i < total) {
    float v = fmaxf(h[i], 0.0f);
    h[i] = v;
    hb[i] = f2bf(v);
  }
}

// ---------------- mean pool ----------------
__global__ __launch_bounds__(256)
void pool_accum_kernel(const float* __restrict__ h, const long long* __restrict__ batch,
                       float* __restrict__ pooled, float* __restrict__ cnts, long long N) {
  long long t = (long long)blockIdx.x * blockDim.x + threadIdx.x;
  long long nidx = t >> 5;
  if (nidx >= N) return;
  int lane = (int)(t & 31);
  long long g = batch[nidx];
  const float4 v = ((const float4*)(h + nidx * D))[lane];
  float* pr = pooled + g * D + lane * 4;
  unsafeAtomicAdd(pr + 0, v.x);
  unsafeAtomicAdd(pr + 1, v.y);
  unsafeAtomicAdd(pr + 2, v.z);
  unsafeAtomicAdd(pr + 3, v.w);
  if (lane == 0) unsafeAtomicAdd(&cnts[g], 1.0f);
}

__global__ void pool_div_kernel(float* pooled, const float* __restrict__ cnts, long long total) {
  long long i = (long long)blockIdx.x * blockDim.x + threadIdx.x;
  if (i < total) pooled[i] /= fmaxf(cnts[i >> 7], 1.0f);
}

__global__ void bias_relu_kernel(float* z, const float* __restrict__ b, long long total) {
  long long i = (long long)blockIdx.x * blockDim.x + threadIdx.x;
  if (i < total) z[i] = fmaxf(z[i] + b[(int)(i & (D - 1))], 0.0f);
}

// ---------------- head: logits = z @ lW2 + lb2 ; out = sigmoid ----------------
__global__ void head2_kernel(const float* __restrict__ z, const float* __restrict__ lW2,
                             const float* __restrict__ lb2, float* __restrict__ out, int G) {
  int g = blockIdx.x * blockDim.x + threadIdx.x;
  if (g >= G) return;
  float acc = lb2[0];
  const float* zr = z + (long long)g * D;
#pragma unroll 8
  for (int k = 0; k < D; ++k) acc += zr[k] * lW2[k];
  out[g] = 1.0f / (1.0f + __expf(-acc));
}

extern "C" void kernel_launch(void* const* d_in, const int* in_sizes, int n_in,
                              void* d_out, int out_size, void* d_ws, size_t ws_size,
                              hipStream_t stream) {
  const float*     x     = (const float*)d_in[0];
  const long long* eidx  = (const long long*)d_in[1];
  const long long* batch = (const long long*)d_in[2];
  const float* W1  = (const float*)d_in[3];
  const float* b1  = (const float*)d_in[4];
  const float* W2  = (const float*)d_in[5];
  const float* b2  = (const float*)d_in[6];
  const float* W3  = (const float*)d_in[7];
  const float* b3  = (const float*)d_in[8];
  const float* lW1 = (const float*)d_in[9];
  const float* lb1 = (const float*)d_in[10];
  const float* lW2 = (const float*)d_in[11];
  const float* lb2 = (const float*)d_in[12];

  const long long N = (long long)in_sizes[0] / D;
  const long long E = (long long)in_sizes[1] / 2;
  const int G = out_size;

  const long long* src = eidx;
  const long long* dst = eidx + E;

  // ---- workspace layout (256B aligned slabs) ----
  char* ws = (char*)d_ws;
  size_t ofs = 0;
  auto alloc = [&](size_t bytes) { void* p = ws + ofs; ofs += (bytes + 255) & ~(size_t)255; return p; };
  float*  xw     = (float*)alloc((size_t)N * D * 4);
  float*  h_a    = (float*)alloc((size_t)N * D * 4);
  float*  h_b    = (float*)alloc((size_t)N * D * 4);
  __bf16* actbf  = (__bf16*)alloc((size_t)N * D * 2);   // bf16 shadow of current GEMM input
  float*  deg    = (float*)alloc((size_t)N * 4);
  float*  dis    = (float*)alloc((size_t)N * 4);
  float*  dinv   = (float*)alloc((size_t)N * 4);
  __bf16* W1p    = (__bf16*)alloc((size_t)D * D * 2);
  __bf16* W2p    = (__bf16*)alloc((size_t)D * D * 2);
  __bf16* W3p    = (__bf16*)alloc((size_t)D * D * 2);
  __bf16* lW1p   = (__bf16*)alloc((size_t)D * D * 2);
  float*  pooled = (float*)alloc((size_t)G * D * 4);
  float*  cnts   = (float*)alloc((size_t)G * 4);
  float*  zbuf   = (float*)alloc((size_t)G * D * 4);

  const int TB = 256;
  const unsigned gN    = (unsigned)((N + TB - 1) / TB);
  const unsigned gE    = (unsigned)((E + TB - 1) / TB);
  const unsigned gNd   = (unsigned)((N * D + TB - 1) / TB);
  const unsigned gE32  = (unsigned)((E * 32 + TB - 1) / TB);
  const unsigned gN32  = (unsigned)((N * 32 + TB - 1) / TB);
  const unsigned gGemm = (unsigned)((N + 15) / 16);
  const unsigned gGemH = (unsigned)((G + 15) / 16);
  const unsigned gW    = (unsigned)((D * D + TB - 1) / TB);
  const unsigned gGd   = (unsigned)(((long long)G * D + TB - 1) / TB);

  // degrees, norms
  init_deg_kernel<<<gN, TB, 0, stream>>>(deg, N);
  deg_accum_kernel<<<gE, TB, 0, stream>>>(dst, deg, E);
  deg_fin_kernel<<<gN, TB, 0, stream>>>(deg, dis, dinv, N);

  // pack weights into WMMA-B lane layout (bf16)
  pack_w_kernel<<<gW, TB, 0, stream>>>(W1,  W1p);
  pack_w_kernel<<<gW, TB, 0, stream>>>(W2,  W2p);
  pack_w_kernel<<<gW, TB, 0, stream>>>(W3,  W3p);
  pack_w_kernel<<<gW, TB, 0, stream>>>(lW1, lW1p);

  // layer 1: x -> h_a
  f32_to_bf16_kernel<<<gNd, TB, 0, stream>>>(x, actbf, N * D);
  wmma_gemm_kernel<<<gGemm, TB, 0, stream>>>(actbf, W1p, xw, N);
  init_agg_kernel<<<gNd, TB, 0, stream>>>(xw, dinv, b1, h_a, N * D);
  scatter_kernel<<<gE32, TB, 0, stream>>>(xw, src, dst, dis, h_a, E);
  relu_bf_kernel<<<gNd, TB, 0, stream>>>(h_a, actbf, N * D);

  // layer 2: h_a -> h_b
  wmma_gemm_kernel<<<gGemm, TB, 0, stream>>>(actbf, W2p, xw, N);
  init_agg_kernel<<<gNd, TB, 0, stream>>>(xw, dinv, b2, h_b, N * D);
  scatter_kernel<<<gE32, TB, 0, stream>>>(xw, src, dst, dis, h_b, E);
  relu_bf_kernel<<<gNd, TB, 0, stream>>>(h_b, actbf, N * D);

  // layer 3: h_b -> h_a
  wmma_gemm_kernel<<<gGemm, TB, 0, stream>>>(actbf, W3p, xw, N);
  init_agg_kernel<<<gNd, TB, 0, stream>>>(xw, dinv, b3, h_a, N * D);
  scatter_kernel<<<gE32, TB, 0, stream>>>(xw, src, dst, dis, h_a, E);
  relu_bf_kernel<<<gNd, TB, 0, stream>>>(h_a, actbf, N * D);

  // mean pool (uses f32 h_a)
  hipMemsetAsync(pooled, 0, (size_t)G * D * 4, stream);
  hipMemsetAsync(cnts, 0, (size_t)G * 4, stream);
  pool_accum_kernel<<<gN32, TB, 0, stream>>>(h_a, batch, pooled, cnts, N);
  pool_div_kernel<<<gGd, TB, 0, stream>>>(pooled, cnts, (long long)G * D);

  // head: pooled -> bf16 -> WMMA GEMM -> bias+relu -> dot+sigmoid
  f32_to_bf16_kernel<<<gGd, TB, 0, stream>>>(pooled, actbf, (long long)G * D);
  wmma_gemm_kernel<<<gGemH, TB, 0, stream>>>(actbf, lW1p, zbuf, G);
  bias_relu_kernel<<<gGd, TB, 0, stream>>>(zbuf, lb1, (long long)G * D);
  head2_kernel<<<(unsigned)((G + TB - 1) / TB), TB, 0, stream>>>(zbuf, lW2, lb2, (float*)d_out, G);
}
